// ConcatCritic_3736621548237
// MI455X (gfx1250) — compile-verified, matrix operand order checked by reading, not run on hardware
//
#include <hip/hip_runtime.h>
#include <stdint.h>
#include <stddef.h>

// ---------------------------------------------------------------------------
// ConcatCritic fused MLP for MI455X (gfx1250, wave32, WMMA bf16).
//
// scores[i,j] = W3 . relu(W2 . relu(W1 . relu(hx[i]+hy[j]+b0) + b1) + b2) + b3
// Dominant cost: 2 GEMMs [65536,512]x[512,512] (68.7 GFLOP) -> compute bound.
// Full fusion through LDS; bf16 operands, fp32 WMMA accumulation.
//
// Wave tiling (v2): each of the 8 waves owns all M=64 rows x a private
// N=64 strip -> no duplicated weight-fragment loads across waves; per k-step
// a wave issues 8 ds_load_b128 (A) + 8 global_load_b128 (B, L2-hot) and
// 16 v_wmma_f32_16x16x32_bf16, with all loads batched before the WMMA burst.
// ---------------------------------------------------------------------------

typedef __attribute__((ext_vector_type(16))) __bf16 v16bf;
typedef __attribute__((ext_vector_type(8)))  float  v8f;

struct alignas(16) U4 { unsigned int x, y, z, w; };
struct alignas(16) F4 { float x, y, z, w; };
struct alignas(32) Frag { U4 lo, hi; };

#define HIDN 512
#define BSZ  256
#define MT   64      // (i,j) rows per workgroup
#define LSTR 520     // LDS row stride in bf16 elems: 512 + 8 pad -> 1040 B
                     // (16B aligned rows; 4-bank rotation/row -> conflict-free
                     //  ds_load_b128 A-fragment loads across 16 rows)

__device__ __forceinline__ unsigned short f2bf(float f) {
  unsigned int u = __float_as_uint(f);
  u += 0x7FFFu + ((u >> 16) & 1u);           // round-to-nearest-even
  return (unsigned short)(u >> 16);
}
__device__ __forceinline__ float bf2f(unsigned short h) {
  return __uint_as_float(((unsigned int)h) << 16);
}

// Load one 16x32 bf16 WMMA fragment (A-style; B uses same layout on W^T).
// ISA layout: lanes 0-15 hold K = {kb..kb+7, kb+16..kb+23} with kb = k0;
// lanes 16-31 hold K = {kb+8..kb+15, kb+24..kb+31}. Row/col = lane & 15.
__device__ __forceinline__ v16bf load_frag(const unsigned short* base) {
  Frag fr;
  fr.lo = *(const U4*)(base);
  fr.hi = *(const U4*)(base + 16);
  return __builtin_bit_cast(v16bf, fr);
}

// One MLP layer: sOut = relu(sIn @ Wt^T + bias), all 64x512, bf16 in LDS.
// Wt is [n][k] bf16 (transposed weights), K-contiguous for fragment loads.
__device__ __forceinline__ void mlp_layer(const unsigned short* sIn,
                                          unsigned short*       sOut,
                                          const unsigned short* __restrict__ Wt,
                                          const float* __restrict__ bias) {
  const int lane = threadIdx.x & 31;
  const int wave = threadIdx.x >> 5;
  const int lr   = lane & 15;           // row (A) / col (B,C) within tile
  const int kh   = (lane >> 4) << 3;    // K sub-offset: 0 or 8
  const int n0   = wave * 64;           // private 64-col strip per wave

  v8f acc[4][4];                        // [tm][tn], 16x16 tiles, 128 VGPRs
#pragma unroll
  for (int a = 0; a < 4; ++a)
#pragma unroll
    for (int b = 0; b < 4; ++b)
      acc[a][b] = (v8f){0.f, 0.f, 0.f, 0.f, 0.f, 0.f, 0.f, 0.f};

  for (int kk = 0; kk < HIDN; kk += 32) {
    const int kb = kk + kh;
    // Batch all fragment loads for this k-step, then burst 16 WMMAs.
    v16bf af[4], bf[4];
#pragma unroll
    for (int tm = 0; tm < 4; ++tm)
      af[tm] = load_frag(&sIn[(tm * 16 + lr) * LSTR + kb]);
#pragma unroll
    for (int tn = 0; tn < 4; ++tn)
      bf[tn] = load_frag(&Wt[(size_t)(n0 + tn * 16 + lr) * HIDN + kb]);
#pragma unroll
    for (int tn = 0; tn < 4; ++tn)
#pragma unroll
      for (int tm = 0; tm < 4; ++tm)
        acc[tm][tn] = __builtin_amdgcn_wmma_f32_16x16x32_bf16(
            false, af[tm], false, bf[tn], (short)0, acc[tm][tn], false, false);
  }

  // Epilogue: bias + relu + bf16 store back to LDS.
  // C layout: lane holds col n = lane&15; VGPR r holds row r + 8*(lane>=16).
  const int mrb = (lane >> 4) << 3;
#pragma unroll
  for (int tn = 0; tn < 4; ++tn) {
    const int n  = n0 + tn * 16 + lr;
    const float bn = bias[n];
#pragma unroll
    for (int tm = 0; tm < 4; ++tm) {
#pragma unroll
      for (int r = 0; r < 8; ++r) {
        float v = fmaxf(acc[tm][tn][r] + bn, 0.f);
        sOut[(tm * 16 + mrb + r) * LSTR + n] = f2bf(v);
      }
    }
  }
  __syncthreads();
}

__global__ __launch_bounds__(256) void fused_critic(
    const float* __restrict__ hx,           // [256][512] fp32, b0 folded in
    const float* __restrict__ hy,           // [256][512] fp32
    const unsigned short* __restrict__ W1t, // [512][512] bf16 [n][k]
    const float* __restrict__ b1,
    const unsigned short* __restrict__ W2t, // [512][512] bf16 [n][k]
    const float* __restrict__ b2,
    const float* __restrict__ W3,           // [512] fp32
    const float* __restrict__ b3,           // [1]  fp32
    float* __restrict__ out)                // [256][256]
{
  extern __shared__ unsigned short smem[];  // 2 x 64 x 520 bf16 = 133,120 B
  unsigned short* sA = smem;
  unsigned short* sB = smem + MT * LSTR;

  const int blk = blockIdx.x;               // 0..1023
  const int i   = blk >> 2;                 // x-row (constant per block)
  const int j0  = (blk & 3) * MT;           // y-row base

  // Phase A: h0 = relu(hx[i] + hy[j]) -> bf16 in sA. 4 threads per row.
  {
    const int t  = threadIdx.x;
    const int r  = t >> 2;
    const int k0 = (t & 3) * 128;
    const F4* px = (const F4*)&hx[(size_t)i * HIDN + k0];
    const F4* py = (const F4*)&hy[(size_t)(j0 + r) * HIDN + k0];
    unsigned short* dst = &sA[r * LSTR + k0];
#pragma unroll 4
    for (int q = 0; q < 32; ++q) {
      F4 a = px[q];
      F4 b = py[q];
      unsigned int p0 = (unsigned int)f2bf(fmaxf(a.x + b.x, 0.f)) |
                        ((unsigned int)f2bf(fmaxf(a.y + b.y, 0.f)) << 16);
      unsigned int p1 = (unsigned int)f2bf(fmaxf(a.z + b.z, 0.f)) |
                        ((unsigned int)f2bf(fmaxf(a.w + b.w, 0.f)) << 16);
      unsigned int* d32 = (unsigned int*)(dst + q * 4);
      d32[0] = p0;
      d32[1] = p1;
    }
  }
  __syncthreads();

  mlp_layer(sA, sB, W1t, b1);   // h1 = relu(h0 @ W1 + b1)  (sync inside)
  mlp_layer(sB, sA, W2t, b2);   // h2 = relu(h1 @ W2 + b2)

  // Phase D: scores = h2 @ W3 + b3. 4-lane partial dots + shfl reduce.
  {
    const int t  = threadIdx.x;
    const int r  = t >> 2;
    const int k0 = (t & 3) * 128;
    const unsigned short* row = &sA[r * LSTR];
    float acc = 0.f;
    for (int k = k0; k < k0 + 128; ++k)
      acc += bf2f(row[k]) * W3[k];
    acc += __shfl_xor(acc, 1, 32);
    acc += __shfl_xor(acc, 2, 32);
    if ((t & 3) == 0)
      out[(size_t)i * BSZ + j0 + r] = acc + b3[0];
  }
}

// Prep 1: hx = x@Wx + b0, hy = y@Wy (fp32). 33 MFLOP total -> negligible.
__global__ __launch_bounds__(256) void prep_proj(
    const float* __restrict__ x,  const float* __restrict__ y,
    const float* __restrict__ Wx, const float* __restrict__ Wy,
    const float* __restrict__ b0,
    float* __restrict__ hx, float* __restrict__ hy)
{
  const int id   = blockIdx.x * 256 + threadIdx.x;   // 0..262143
  const int half = id >> 17;
  const int idx  = id & 131071;
  const int row  = idx >> 9;
  const int h    = idx & 511;
  if (half == 0) {
    float acc = b0[h];
    for (int k = 0; k < 128; ++k) acc += x[row * 128 + k] * Wx[k * 512 + h];
    hx[idx] = acc;
  } else {
    float acc = 0.f;
    for (int k = 0; k < 128; ++k) acc += y[row * 128 + k] * Wy[k * 512 + h];
    hy[idx] = acc;
  }
}

// Prep 2: pack W1, W2 as transposed bf16 [n][k] for K-contiguous B fragments.
__global__ __launch_bounds__(256) void prep_pack(
    const float* __restrict__ W1, const float* __restrict__ W2,
    unsigned short* __restrict__ W1t, unsigned short* __restrict__ W2t)
{
  const int id   = blockIdx.x * 256 + threadIdx.x;   // 0..524287
  const int half = id >> 18;
  const int idx  = id & 262143;
  const int n    = idx >> 9;
  const int k    = idx & 511;
  const float* W = half ? W2 : W1;
  unsigned short* Wt = half ? W2t : W1t;
  Wt[idx] = f2bf(W[k * 512 + n]);
}

extern "C" void kernel_launch(void* const* d_in, const int* in_sizes, int n_in,
                              void* d_out, int out_size, void* d_ws, size_t ws_size,
                              hipStream_t stream) {
  const float* x  = (const float*)d_in[0];
  const float* y  = (const float*)d_in[1];
  const float* Wx = (const float*)d_in[2];
  const float* Wy = (const float*)d_in[3];
  const float* b0 = (const float*)d_in[4];
  const float* W1 = (const float*)d_in[5];
  const float* b1 = (const float*)d_in[6];
  const float* W2 = (const float*)d_in[7];
  const float* b2 = (const float*)d_in[8];
  const float* W3 = (const float*)d_in[9];
  const float* b3 = (const float*)d_in[10];
  float* out = (float*)d_out;

  // Workspace layout (2.0 MB total; all L2-resident on a 192 MB L2):
  float* hx = (float*)d_ws;                       // 256*512 f32 = 512 KB
  float* hy = hx + 256 * 512;                     // 512 KB
  unsigned short* W1t = (unsigned short*)(hy + 256 * 512);  // 512 KB bf16
  unsigned short* W2t = W1t + 512 * 512;                    // 512 KB bf16

  prep_proj<<<1024, 256, 0, stream>>>(x, y, Wx, Wy, b0, hx, hy);
  prep_pack<<<2048, 256, 0, stream>>>(W1, W2, W1t, W2t);

  const size_t smem_bytes = 2u * MT * LSTR * sizeof(unsigned short); // 133,120
  fused_critic<<<1024, 256, smem_bytes, stream>>>(hx, hy, W1t, b1, W2t, b2,
                                                  W3, b3, out);
}
